// OpenMaxClassifier_13365938225589
// MI455X (gfx1250) — matrix operand order repaired and checked by reading, not compiled, not run on hardware
//
#include <hip/hip_runtime.h>

typedef __attribute__((ext_vector_type(16))) _Float16 v16h;
typedef __attribute__((ext_vector_type(8)))  float    v8f;

#define NROWS 262144
#define DDIM  128
#define NCLS  100
#define CPAD  112           // 7 tiles of 16
#define LDH   136           // padded LDS row stride in halfs (avoids bank conflicts)
#define WAVES 8
#define ROWS_PER_WG (WAVES * 16)   // 128 rows per block

#define LOG2E 1.4426950408889634f

union Pack4 { _Float16 h[4]; uint2 u; };
union Frag  { uint4 u[2]; v16h v; };

// Build a 16-half fragment from two 16-byte LDS runs (row-local columns c0, c1).
__device__ __forceinline__ v16h make_frag(const _Float16* row, int c0, int c1) {
    Frag f;
    f.u[0] = *(const uint4*)(row + c0);
    f.u[1] = *(const uint4*)(row + c1);
    return f.v;
}

__global__ __launch_bounds__(256)
void openmax_wmma_kernel(const float* __restrict__ x,
                         const float* __restrict__ mavs,
                         const float* __restrict__ wb_shape,
                         const float* __restrict__ wb_loc,
                         const float* __restrict__ wb_scale,
                         float* __restrict__ out) {
    __shared__ __align__(16) _Float16 smavs[CPAD * LDH];       // 30464 B
    __shared__ __align__(16) _Float16 sx[WAVES][16 * LDH];     // 34816 B
    __shared__ float sm2[CPAD], skk[CPAD], sloc[CPAD], sklg[CPAD];

    const int tid  = threadIdx.x;
    const int lane = tid & 31;
    const int w    = tid >> 5;        // wave id within block
    const int grp  = lane >> 4;       // half-wave: 0 or 1
    const int lr   = lane & 15;

    // ---- Phase 1a: mavs (f32 global) -> f16 LDS, whole block ----
    for (int t = tid; t < (NCLS * DDIM) / 4; t += 256) {       // 3200 float4
        float4 f = ((const float4*)mavs)[t];
        int row = t >> 5;             // (4t)/128
        int col = (t & 31) << 2;
        Pack4 p;
        p.h[0] = (_Float16)f.x; p.h[1] = (_Float16)f.y;
        p.h[2] = (_Float16)f.z; p.h[3] = (_Float16)f.w;
        *(uint2*)&smavs[row * LDH + col] = p.u;
    }
    // Weibull params. Precompute k*log2(scale) so the hot loop needs no
    // divide: ((d-loc)/scale)^k = exp2(k*log2(d-loc) - k*log2(scale)).
    if (tid < CPAD) {
        bool v = tid < NCLS;
        float kk = v ? wb_shape[tid] : 1.0f;
        float sc = v ? wb_scale[tid] : 1.0f;
        skk[tid]  = kk;
        sloc[tid] = v ? wb_loc[tid] : 0.0f;
        sklg[tid] = kk * __builtin_amdgcn_logf(sc);
    }

    // ---- Phase 1b: per-wave x tile (16 rows, contiguous) -> f16 LDS ----
    const int row0 = blockIdx.x * ROWS_PER_WG + w * 16;
    const float* xt = x + (size_t)row0 * DDIM;
    _Float16* sxw = sx[w];
    #pragma unroll
    for (int i = 0; i < 16; ++i) {                             // 512 float4 per tile
        float4 f = ((const float4*)xt)[i * 32 + lane];
        Pack4 p;
        p.h[0] = (_Float16)f.x; p.h[1] = (_Float16)f.y;
        p.h[2] = (_Float16)f.z; p.h[3] = (_Float16)f.w;
        *(uint2*)&sxw[i * LDH + (lane << 2)] = p.u;
    }
    __syncthreads();

    // ---- Phase 2a: m2 from the f16 copies (consistency with WMMA dot) ----
    if (tid < CPAD) {
        float s = 0.0f;
        if (tid < NCLS) {
            const _Float16* r = &smavs[tid * LDH];
            for (int k = 0; k < DDIM; ++k) { float h = (float)r[k]; s += h * h; }
        }
        sm2[tid] = s;
    }

    // ---- Phase 2b: x2 per row via half-row partials + wave32 shuffles ----
    float x2p = 0.0f;
    {
        const _Float16* r = &sxw[lr * LDH + grp * 64];
        for (int k = 0; k < 64; ++k) { float h = (float)r[k]; x2p += h * h; }
    }
    x2p += __shfl_xor(x2p, 16);       // lane l now holds x2[row l&15]
    float x2v[8];
    #pragma unroll
    for (int v = 0; v < 8; ++v) x2v[v] = __shfl(x2p, v + 8 * grp); // x2 of row M=v+8*grp
    __syncthreads();                  // sm2 ready

    // ---- Preload A fragments (K = 128 -> 4 chunks of 32) ----
    // A layout (16-bit, 16x32): lane = M; VGPR0-3 K = grp*8..grp*8+7, VGPR4-7 +16.
    v16h afrag[4];
    #pragma unroll
    for (int kc = 0; kc < 4; ++kc) {
        int c0 = kc * 32 + grp * 8;
        afrag[kc] = make_frag(&sxw[lr * LDH], c0, c0 + 16);
    }

    float rowsum[8] = {0.f, 0.f, 0.f, 0.f, 0.f, 0.f, 0.f, 0.f};

    // ---- Main loop: 7 class tiles, 4 WMMA each ----
    for (int ct = 0; ct < 7; ++ct) {
        const _Float16* brow = &smavs[(ct * 16 + lr) * LDH];
        v8f acc = {};
        #pragma unroll
        for (int kc = 0; kc < 4; ++kc) {
            // B layout (32x16): lane = N; VGPR0-3 K = grp*8..grp*8+7, VGPR4-7 +16.
            int c0 = kc * 32 + grp * 8;
            v16h b = make_frag(brow, c0, c0 + 16);
            acc = __builtin_amdgcn_wmma_f32_16x16x32_f16(
                false, afrag[kc], false, b, (short)0, acc, false, false);
        }

        const int  c     = ct * 16 + lr;
        const bool valid = c < NCLS;
        const float m2c  = sm2[c];
        const float kk   = skk[c];
        const float lc   = sloc[c];
        const float kl   = sklg[c];

        #pragma unroll
        for (int v = 0; v < 8; ++v) {
            // D element (M = v + 8*grp, N = lr)
            float sq  = fmaf(-2.0f, acc[v], x2v[v] + m2c);
            float d   = __builtin_amdgcn_sqrtf(fmaxf(sq, 1e-12f));
            float t   = d - lc;
            float lg  = __builtin_amdgcn_logf(t);                 // log2(d - loc)
            float pw  = __builtin_amdgcn_exp2f(fmaf(kk, lg, -kl)); // ((d-loc)/scale)^k
            pw = (t > 0.0f) ? pw : 0.0f;
            float cdf = 1.0f - __builtin_amdgcn_exp2f(-LOG2E * pw);
            float om  = __builtin_amdgcn_exp2f(-LOG2E * d) * cdf;
            if (valid) {
                int row = row0 + v + 8 * grp;
                out[(size_t)row * (NCLS + 1) + c] = om;
                rowsum[v] += om;
            }
        }
    }

    // ---- Unknown column: reduce openmax over classes (16-lane reduction) ----
    #pragma unroll
    for (int v = 0; v < 8; ++v) {
        float s = rowsum[v];
        s += __shfl_xor(s, 1);
        s += __shfl_xor(s, 2);
        s += __shfl_xor(s, 4);
        s += __shfl_xor(s, 8);
        if (lr == 0) {
            int row = row0 + v + 8 * grp;
            out[(size_t)row * (NCLS + 1) + NCLS] = 1.0f - s;
        }
    }
}

extern "C" void kernel_launch(void* const* d_in, const int* in_sizes, int n_in,
                              void* d_out, int out_size, void* d_ws, size_t ws_size,
                              hipStream_t stream) {
    (void)in_sizes; (void)n_in; (void)out_size; (void)d_ws; (void)ws_size;
    const float* x     = (const float*)d_in[0];
    const float* mavs  = (const float*)d_in[1];
    const float* wbsh  = (const float*)d_in[2];
    const float* wblo  = (const float*)d_in[3];
    const float* wbsc  = (const float*)d_in[4];
    float* out = (float*)d_out;

    dim3 grid(NROWS / ROWS_PER_WG);   // 2048 blocks
    dim3 block(256);                  // 8 wave32 waves
    hipLaunchKernelGGL(openmax_wmma_kernel, grid, block, 0, stream,
                       x, mavs, wbsh, wblo, wbsc, out);
}
